// Epipolar_Attention_23407571763461
// MI455X (gfx1250) — matrix operand order repaired and checked by reading, not compiled, not run on hardware
//
#include <hip/hip_runtime.h>
#include <math.h>

// ---------------------------------------------------------------------------
// Types for gfx1250 WMMA (wave32): D(f32 16x16) = A(bf16 16x32) x B(bf16 32x16) + C
// ---------------------------------------------------------------------------
typedef __attribute__((ext_vector_type(16))) __bf16        v16bf;
typedef __attribute__((ext_vector_type(8)))  float         v8f;
typedef __attribute__((ext_vector_type(8)))  unsigned int  v8u;
typedef __attribute__((ext_vector_type(4)))  unsigned int  v4ui;
typedef __attribute__((ext_vector_type(8)))  int           v8i;
typedef __attribute__((ext_vector_type(4)))  int           v4i;

union Frag { v16bf bf; v8u u; };

__device__ __forceinline__ unsigned short f2bf(float f) {
    unsigned int u = __float_as_uint(f);
    u += 0x7fffu + ((u >> 16) & 1u);          // round-to-nearest-even
    return (unsigned short)(u >> 16);
}

// A-fragment (16x32 bf16): lane&15 = M row; element j -> K = (j>>3)*16 + (lane>>4)*8 + (j&7)
__device__ __forceinline__ v16bf load_fragA(const unsigned int* __restrict__ base,
                                            int row, int ld, int kbase, int h8) {
    Frag f;
#pragma unroll
    for (int r = 0; r < 8; ++r) {
        int k = kbase + ((r >> 2) << 4) + h8 + ((r & 3) << 1);
        f.u[r] = base[(row * ld + k) >> 1];
    }
    return f.bf;
}

// B-fragment (32x16 bf16): lane&15 = N col; element j -> K = (j>>3)*16 + (lane>>4)*8 + (j&7)
__device__ __forceinline__ v16bf load_fragB(const unsigned int* __restrict__ base,
                                            int kbase, int ld, int n, int h8) {
    Frag f;
    const unsigned short* p = (const unsigned short*)base;
#pragma unroll
    for (int r = 0; r < 8; ++r) {
        int k = kbase + ((r >> 2) << 4) + h8 + ((r & 3) << 1);
        f.u[r] = (unsigned int)p[(size_t)k * ld + n] | ((unsigned int)p[(size_t)(k + 1) * ld + n] << 16);
    }
    return f.bf;
}

__device__ __forceinline__ v16bf load_fragB_lds(const unsigned short* p, int ld, int h8) {
    Frag f;
#pragma unroll
    for (int r = 0; r < 8; ++r) {
        int k = ((r >> 2) << 4) + h8 + ((r & 3) << 1);
        f.u[r] = (unsigned int)p[k * ld] | ((unsigned int)p[(k + 1) * ld] << 16);
    }
    return f.bf;
}

__device__ __forceinline__ v8f wmma_bf16(v16bf a, v16bf b, v8f c) {
    return __builtin_amdgcn_wmma_f32_16x16x32_bf16(false, a, false, b, (short)0, c, false, false);
}

// ---------------------------------------------------------------------------
// Tensor Data Mover: 2D tile load global->LDS (D# packed per CDNA5 ISA 8.3/8.4)
// This toolchain exposes the 6-arg builtin: (v4u g0, v8i g1, v4i g2, v4i g3, v8i g4, i32 cpol)
// ---------------------------------------------------------------------------
typedef __attribute__((address_space(3))) const void as3_cvoid;

__device__ __forceinline__ unsigned lds_offset(const void* p) {
    return (unsigned)(unsigned long long)(as3_cvoid*)p;   // AS3 ptr = 32-bit LDS byte offset
}

// 2-byte elements; tile_w contiguous along dim0; row stride in elements.
__device__ __forceinline__ void tdm_load_2d(unsigned lds_addr, const void* gsrc,
                                            unsigned tile_w, unsigned tile_h,
                                            unsigned tensor_w, unsigned tensor_h,
                                            unsigned row_stride)
{
    unsigned long long ga = (unsigned long long)gsrc;
    v4ui g0;
    g0[0] = 1u;                                            // count=1 (valid user D#)
    g0[1] = lds_addr;                                      // lds_addr (bytes)
    g0[2] = (unsigned)(ga & 0xffffffffu);                  // global_addr[31:0]
    g0[3] = (unsigned)((ga >> 32) & 0x1ffffffu) | (2u << 30); // global_addr[56:32] | type=2
    v8i g1;
    g1[0] = (int)(1u << 16);                               // data_size=1 -> 2 bytes
    g1[1] = (int)((tensor_w & 0xffffu) << 16);             // tensor_dim0[15:0] @ bits63:48
    g1[2] = (int)(((tensor_w >> 16) & 0xffffu) | ((tensor_h & 0xffffu) << 16));
    g1[3] = (int)(((tensor_h >> 16) & 0xffffu) | ((tile_w & 0xffffu) << 16)); // tile_dim0
    g1[4] = (int)(tile_h & 0xffffu);                       // tile_dim1 (tile_dim2=0)
    g1[5] = (int)row_stride;                               // tensor_dim0_stride[31:0]
    g1[6] = 0;
    g1[7] = 0;
    v4i z4 = {0, 0, 0, 0};
    v8i z8 = {0, 0, 0, 0, 0, 0, 0, 0};
    __builtin_amdgcn_tensor_load_to_lds(g0, g1, z4, z4, z8, 0);
}

// ---------------------------------------------------------------------------
// Epipolar geometry: per (dir,b) precompute projected points pi[n] + epipole oi
// ---------------------------------------------------------------------------
__global__ void epi_prep_kernel(const float* __restrict__ intr, const float* __restrict__ c2w,
                                float* __restrict__ pi, float* __restrict__ oi)
{
    int dir = blockIdx.x >> 2;          // 0: src=c2w[1],tgt=c2w[0]   1: src=c2w[0],tgt=c2w[1]
    int b   = blockIdx.x & 3;
    int srcSel = (dir == 0) ? 1 : 0;
    int tgtSel = 1 - srcSel;

    const float* Ki = intr + b * 9;
    const float Wimg = 32.0f * 16.0f / 9.0f;
    float K00 = Ki[0] * Wimg, K01 = Ki[1] * Wimg, K02 = 16.0f;
    float K10 = Ki[3] * 32.0f, K11 = Ki[4] * 32.0f, K12 = 16.0f;
    float K20 = Ki[6],        K21 = Ki[7],          K22 = Ki[8];

    const float* S = c2w + (size_t)(srcSel * 4 + b) * 16;
    const float* T = c2w + (size_t)(tgtSel * 4 + b) * 16;
    float sr0 = S[0], sr1 = S[1], sr2 = S[2];
    float sr3 = S[4], sr4 = S[5], sr5 = S[6];
    float sr6 = S[8], sr7 = S[9], sr8 = S[10];
    float st0 = S[3], st1 = S[7], st2 = S[11];

    float a00 = T[0], a01 = T[1], a02 = T[2];
    float a10 = T[4], a11 = T[5], a12 = T[6];
    float a20 = T[8], a21 = T[9], a22 = T[10];
    float det = a00 * (a11 * a22 - a12 * a21) - a01 * (a10 * a22 - a12 * a20)
              + a02 * (a10 * a21 - a11 * a20);
    float id = 1.0f / det;
    float r00 =  (a11 * a22 - a12 * a21) * id, r01 = -(a01 * a22 - a02 * a21) * id, r02 =  (a01 * a12 - a02 * a11) * id;
    float r10 = -(a10 * a22 - a12 * a20) * id, r11 =  (a00 * a22 - a02 * a20) * id, r12 = -(a00 * a12 - a02 * a10) * id;
    float r20 =  (a10 * a21 - a11 * a20) * id, r21 = -(a00 * a21 - a01 * a20) * id, r22 =  (a00 * a11 - a01 * a10) * id;
    float tt0 = -T[3], tt1 = -T[7], tt2 = -T[11];

    float ox = r00 * st0 + r01 * st1 + r02 * st2 + tt0;
    float oy = r10 * st0 + r11 * st1 + r12 * st2 + tt1;
    float oz = r20 * st0 + r21 * st1 + r22 * st2 + tt2;
    float kx = K00 * ox + K01 * oy + K02 * oz;
    float ky = K10 * ox + K11 * oy + K12 * oz;
    float kz = K20 * ox + K21 * oy + K22 * oz;
    if (threadIdx.x == 0) {
        float* od = oi + (dir * 4 + b) * 4;
        od[0] = kx / kz; od[1] = ky / kz; od[2] = 1.0f;
    }

    for (int n = threadIdx.x; n < 1024; n += blockDim.x) {
        float cx = (float)(n & 31), cy = (float)(n >> 5);
        float camx = (cx - 16.0f) / K00, camy = (cy - 16.0f) / K11;
        float pwx = sr0 * camx + sr1 * camy + sr2 + st0;
        float pwy = sr3 * camx + sr4 * camy + sr5 + st1;
        float pwz = sr6 * camx + sr7 * camy + sr8 + st2;
        float ptx = r00 * pwx + r01 * pwy + r02 * pwz + tt0;
        float pty = r10 * pwx + r11 * pwy + r12 * pwz + tt1;
        float ptz = r20 * pwx + r21 * pwy + r22 * pwz + tt2;
        float px = K00 * ptx + K01 * pty + K02 * ptz;
        float py = K10 * ptx + K11 * pty + K12 * ptz;
        float pz = K20 * ptx + K21 * pty + K22 * ptz;
        float inv = 1.0f / (pz + 1e-6f);
        float* pd = pi + ((size_t)(dir * 4 + b) * 1024 + n) * 4;
        pd[0] = px * inv; pd[1] = py * inv; pd[2] = pz * inv;
    }
}

__global__ void epi_rows_kernel(const float* __restrict__ pi, const float* __restrict__ oi,
                                float* __restrict__ f1, float* __restrict__ f2)
{
    __shared__ float red[256];
    int idx = blockIdx.x;
    int dir = idx >> 12;
    int b   = (idx >> 10) & 3;
    int m   = idx & 1023;
    const float* od = oi + (dir * 4 + b) * 4;
    const float* pd = pi + ((size_t)(dir * 4 + b) * 1024 + m) * 4;
    float ux = pd[0] - od[0], uy = pd[1] - od[1], uz = pd[2] - od[2];
    float vlen = sqrtf(ux * ux + uy * uy + uz * uz);
    int tid = threadIdx.x;
    float dwv[4];
    float lmax = -1.0f;
#pragma unroll
    for (int s = 0; s < 4; ++s) {
        int n = tid + s * 256;
        float wx = (float)(n & 31) - od[0];
        float wy = (float)(n >> 5) - od[1];
        float wz = 1.0f - od[2];
        float cxp = uy * wz - uz * wy;
        float cyp = uz * wx - ux * wz;
        float czp = ux * wy - uy * wx;
        float dist = sqrtf(cxp * cxp + cyp * cyp + czp * czp) / vlen;
        float dw = 1.0f - 1.0f / (1.0f + expf(-50.0f * (dist - 0.5f)));
        dwv[s] = dw;
        lmax = fmaxf(lmax, dw);
    }
    red[tid] = lmax; __syncthreads();
    for (int s = 128; s > 0; s >>= 1) { if (tid < s) red[tid] = fmaxf(red[tid], red[tid + s]); __syncthreads(); }
    bool fill = red[0] < 0.5f;
    float* f = (dir == 0) ? f1 : f2;
    float* out = f + (((size_t)(b * 1024 + m)) << 10);
#pragma unroll
    for (int s = 0; s < 4; ++s) out[tid + s * 256] = fill ? 1.0f : dwv[s];
}

// ---------------------------------------------------------------------------
__global__ void cvt_bf16_kernel(const float* __restrict__ src, unsigned short* __restrict__ dst, int n)
{
    int i = blockIdx.x * 256 + threadIdx.x;
    if (i < n) dst[i] = f2bf(src[i]);
}

// ---------------------------------------------------------------------------
// LayerNorm over C=320. Optional outputs: bf16, f32, transposed-f32 (B,C,HW).
// ---------------------------------------------------------------------------
__global__ void ln_kernel(const float* __restrict__ X, const float* __restrict__ g,
                          const float* __restrict__ bb, unsigned short* Ybf,
                          float* Yf, float* Yt, int in_transposed)
{
    const int Cdim = 320;
    __shared__ float red[256];
    int row = blockIdx.x;
    int bidx = row >> 10, n = row & 1023;
    int tid = threadIdx.x;
    int c0 = tid, c1 = tid + 256;
    bool has1 = c1 < Cdim;
    size_t i0 = in_transposed ? ((size_t)(bidx * Cdim + c0)) * 1024 + n : (size_t)row * Cdim + c0;
    size_t i1 = in_transposed ? ((size_t)(bidx * Cdim + c1)) * 1024 + n : (size_t)row * Cdim + c1;
    float x0 = X[i0];
    float x1 = has1 ? X[i1] : 0.0f;
    red[tid] = x0 + x1; __syncthreads();
    for (int s = 128; s > 0; s >>= 1) { if (tid < s) red[tid] += red[tid + s]; __syncthreads(); }
    float mean = red[0] * (1.0f / Cdim);
    __syncthreads();
    float d0 = x0 - mean, d1 = has1 ? x1 - mean : 0.0f;
    red[tid] = d0 * d0 + d1 * d1; __syncthreads();
    for (int s = 128; s > 0; s >>= 1) { if (tid < s) red[tid] += red[tid + s]; __syncthreads(); }
    float inv = rsqrtf(red[0] * (1.0f / Cdim) + 1e-5f);

    float y0 = d0 * inv * g[c0] + bb[c0];
    size_t o0 = (size_t)row * Cdim + c0;
    if (Ybf) Ybf[o0] = f2bf(y0);
    if (Yf)  Yf[o0] = y0;
    if (Yt)  Yt[((size_t)(bidx * Cdim + c0)) * 1024 + n] = y0;
    if (has1) {
        float y1 = d1 * inv * g[c1] + bb[c1];
        size_t o1 = (size_t)row * Cdim + c1;
        if (Ybf) Ybf[o1] = f2bf(y1);
        if (Yf)  Yf[o1] = y1;
        if (Yt)  Yt[((size_t)(bidx * Cdim + c1)) * 1024 + n] = y1;
    }
}

// ---------------------------------------------------------------------------
// WMMA GEMM: each wave computes a 16x64 strip (A-fragment reused over 4 WMMAs)
// C(MxN,f32) = A(MxK,bf16) @ B(KxN,bf16) + bias; optional GELU / residual / bf16 out
// ---------------------------------------------------------------------------
__global__ __launch_bounds__(128) void gemm_bf16_kernel(
    const unsigned int* __restrict__ A, const unsigned int* __restrict__ Bw,
    const float* __restrict__ bias, const float* __restrict__ residual,
    float* Cf, unsigned short* Cbf, int M, int N, int K, int do_gelu)
{
    int wave = threadIdx.x >> 5, lane = threadIdx.x & 31;
    int tile = blockIdx.x * (blockDim.x >> 5) + wave;
    int tn = N >> 6;                              // 64-wide N strips
    if (tile >= (M >> 4) * tn) return;
    int m0 = (tile / tn) << 4, n0 = (tile % tn) << 6;
    int l15 = lane & 15, h8 = (lane >> 4) << 3;

    v8f acc[4];
#pragma unroll
    for (int j = 0; j < 4; ++j)
#pragma unroll
        for (int r = 0; r < 8; ++r) acc[j][r] = 0.0f;

    int arow = m0 + l15;
    for (int k = 0; k < K; k += 32) {
        v16bf af = load_fragA(A, arow, K, k, h8);
#pragma unroll
        for (int j = 0; j < 4; ++j) {
            v16bf bf = load_fragB(Bw, k, N, n0 + j * 16 + l15, h8);
            acc[j] = wmma_bf16(af, bf, acc[j]);
        }
    }
#pragma unroll
    for (int j = 0; j < 4; ++j) {
        int n = n0 + j * 16 + l15;
        float bs = bias[n];
#pragma unroll
        for (int r = 0; r < 8; ++r) {
            int m = m0 + r + h8;
            float v = acc[j][r] + bs;
            if (do_gelu) v = 0.5f * v * (1.0f + erff(v * 0.70710678118654752f));
            size_t o = (size_t)m * N + n;
            if (residual) v += residual[o];
            if (Cf)  Cf[o]  = v;
            if (Cbf) Cbf[o] = f2bf(v);
        }
    }
}

// ---------------------------------------------------------------------------
// Pack projections into WMMA-friendly bf16 layouts (zero-padded head dim)
//   q: (b,h,1024,64)   A-side, d-contiguous
//   k: (b,h,64,1024)   B-side for QK^T: d rows, key-contiguous
//   v: (b,h,1024,48)   B-side for A@V: key rows, d-contiguous
// ---------------------------------------------------------------------------
__global__ void pack_q_kernel(const float* __restrict__ src, unsigned short* __restrict__ dst)
{
    int idx = blockIdx.x * 256 + threadIdx.x;
    int d = idx & 63, q = (idx >> 6) & 1023, hh = (idx >> 16) & 7, b = idx >> 19;
    float v = (d < 40) ? src[(size_t)(b * 1024 + q) * 320 + hh * 40 + d] : 0.0f;
    dst[idx] = f2bf(v);
}
__global__ void pack_k_kernel(const float* __restrict__ src, unsigned short* __restrict__ dst)
{
    int idx = blockIdx.x * 256 + threadIdx.x;
    int key = idx & 1023, d = (idx >> 10) & 63, hh = (idx >> 16) & 7, b = idx >> 19;
    float v = (d < 40) ? src[(size_t)(b * 1024 + key) * 320 + hh * 40 + d] : 0.0f;
    dst[idx] = f2bf(v);
}
__global__ void pack_v_kernel(const float* __restrict__ src, unsigned short* __restrict__ dst)
{
    int idx = blockIdx.x * 256 + threadIdx.x;
    if (idx >= 4 * 8 * 1024 * 48) return;
    int d = idx % 48; int rest = idx / 48;
    int key = rest & 1023, hh = (rest >> 10) & 7, b = rest >> 13;
    float v = (d < 40) ? src[(size_t)(b * 1024 + key) * 320 + hh * 40 + d] : 0.0f;
    dst[idx] = f2bf(v);
}

// ---------------------------------------------------------------------------
// Fused flash attention with epipolar weighting + TDM double-buffered K/V tiles.
// All 4 waves of a block share (b,head); wave 0 DMAs the 64x32 K-tile and
// 32x48 V-tile for the next key block while everyone computes on the current.
// ---------------------------------------------------------------------------
#define SCALE_F 0.15811388300841897f   // 40^-0.5
__global__ __launch_bounds__(128) void attn_kernel(
    const unsigned int* __restrict__ qp, const unsigned short* __restrict__ kT,
    const unsigned short* __restrict__ vp, const float* __restrict__ f1,
    const float* __restrict__ f2, unsigned short* __restrict__ abf)
{
    __shared__ __align__(16) unsigned short sK[2][64 * 32];   // [d][key_local]
    __shared__ __align__(16) unsigned short sV[2][32 * 48];   // [key_local][d]
    __shared__ __align__(16) unsigned short sP[4 * 16 * 34];  // per-wave P tile, stride 34

    int wave = threadIdx.x >> 5, lane = threadIdx.x & 31;
    int t  = blockIdx.x * 4 + wave;        // 0..2047 ; all waves in block share b,hh
    int b  = t >> 9;
    int hh = (t >> 6) & 7;
    int q0 = (t & 63) << 4;
    int l15 = lane & 15, h8 = (lane >> 4) << 3;

    const unsigned int*   qb  = qp + (size_t)(b * 8 + hh) * 1024 * 32;   // dwords
    const unsigned short* ktb = kT + (size_t)(b * 8 + hh) * 64 * 1024;   // elements
    const unsigned short* vtb = vp + (size_t)(b * 8 + hh) * 1024 * 48;

    v16bf aq0 = load_fragA(qb, q0 + l15, 64, 0,  h8);
    v16bf aq1 = load_fragA(qb, q0 + l15, 64, 32, h8);

    float m_run[8], l_run[8];
    v8f O[3];
#pragma unroll
    for (int r = 0; r < 8; ++r) { m_run[r] = -3.0e38f; l_run[r] = 0.0f; }
#pragma unroll
    for (int t3 = 0; t3 < 3; ++t3)
#pragma unroll
        for (int r = 0; r < 8; ++r) O[t3][r] = 0.0f;

    const int woff = wave * 544;
    unsigned int* sPd = (unsigned int*)sP;

    // prologue: DMA first K/V tiles
    if (wave == 0) {
        tdm_load_2d(lds_offset(&sK[0][0]), ktb,        32, 64, 1024, 64,   1024);
        tdm_load_2d(lds_offset(&sV[0][0]), vtb,        48, 32, 48,   1024, 48);
        __builtin_amdgcn_s_wait_tensorcnt((unsigned short)0);
    }
    __syncthreads();

    for (int kb = 0; kb < 1024; kb += 32) {
        int cur = (kb >> 5) & 1;
        if (wave == 0 && kb + 32 < 1024) {   // prefetch next tiles via TDM
            tdm_load_2d(lds_offset(&sK[cur ^ 1][0]), ktb + (kb + 32),              32, 64, 1024, 64,   1024);
            tdm_load_2d(lds_offset(&sV[cur ^ 1][0]), vtb + (size_t)(kb + 32) * 48, 48, 32, 48,   1024, 48);
        }
        const unsigned int* sKd = (const unsigned int*)&sK[cur][0];
        const unsigned int* sVd = (const unsigned int*)&sV[cur][0];

        float sv[2][8];
#pragma unroll
        for (int nt = 0; nt < 2; ++nt) {
            v8f s;
#pragma unroll
            for (int r = 0; r < 8; ++r) s[r] = 0.0f;
            v16bf b0 = load_fragB_lds(&sK[cur][0] + 0  * 32 + nt * 16 + l15, 32, h8);
            s = wmma_bf16(aq0, b0, s);
            v16bf b1 = load_fragB_lds(&sK[cur][0] + 32 * 32 + nt * 16 + l15, 32, h8);
            s = wmma_bf16(aq1, b1, s);
            int ng = kb + nt * 16 + l15;
#pragma unroll
            for (int r = 0; r < 8; ++r) {
                int mg = q0 + r + h8;
                float e = f1[(((size_t)(b * 1024 + mg)) << 10) + ng]
                        * f2[(((size_t)(b * 1024 + ng)) << 10) + mg];
                sv[nt][r] = s[r] * SCALE_F * e;
            }
        }
        // online softmax over this 32-key block (16-lane row reductions)
#pragma unroll
        for (int r = 0; r < 8; ++r) {
            float mx = fmaxf(sv[0][r], sv[1][r]);
#pragma unroll
            for (int msk = 8; msk >= 1; msk >>= 1) mx = fmaxf(mx, __shfl_xor(mx, msk, 32));
            float mn = fmaxf(m_run[r], mx);
            float alpha = __expf(m_run[r] - mn);
            m_run[r] = mn;
            float p0 = __expf(sv[0][r] - mn), p1 = __expf(sv[1][r] - mn);
            sv[0][r] = p0; sv[1][r] = p1;
            float rs = p0 + p1;
#pragma unroll
            for (int msk = 8; msk >= 1; msk >>= 1) rs += __shfl_xor(rs, msk, 32);
            l_run[r] = l_run[r] * alpha + rs;
#pragma unroll
            for (int t3 = 0; t3 < 3; ++t3) O[t3][r] *= alpha;
        }
        // stage P (C layout) -> LDS -> reload in A-fragment layout
#pragma unroll
        for (int nt = 0; nt < 2; ++nt)
#pragma unroll
            for (int r = 0; r < 8; ++r)
                sP[woff + (r + h8) * 34 + nt * 16 + l15] = f2bf(sv[nt][r]);
        asm volatile("s_wait_dscnt 0" ::: "memory");
        Frag pf;
#pragma unroll
        for (int r = 0; r < 8; ++r) {
            int k = ((r >> 2) << 4) + h8 + ((r & 3) << 1);
            pf.u[r] = sPd[(woff + l15 * 34 + k) >> 1];
        }
#pragma unroll
        for (int t3 = 0; t3 < 3; ++t3) {
            v16bf bv = load_fragB_lds(&sV[cur][0] + t3 * 16 + l15, 48, h8);
            O[t3] = wmma_bf16(pf.bf, bv, O[t3]);
        }
        if (wave == 0) __builtin_amdgcn_s_wait_tensorcnt((unsigned short)0);
        __syncthreads();   // publish next tiles / retire current before overwrite
    }
    // epilogue: O/l -> bf16 (b,hw,320)
#pragma unroll
    for (int t3 = 0; t3 < 3; ++t3) {
        int d = t3 * 16 + l15;
        if (d < 40) {
#pragma unroll
            for (int r = 0; r < 8; ++r) {
                int mg = q0 + r + h8;
                abf[(size_t)(b * 1024 + mg) * 320 + hh * 40 + d] = f2bf(O[t3][r] / l_run[r]);
            }
        }
    }
}

// ---------------------------------------------------------------------------
extern "C" void kernel_launch(void* const* d_in, const int* in_sizes, int n_in,
                              void* d_out, int out_size, void* d_ws, size_t ws_size,
                              hipStream_t stream)
{
    (void)in_sizes; (void)n_in; (void)out_size; (void)ws_size;
    const float* x       = (const float*)d_in[0];
    const float* srcE    = (const float*)d_in[1];
    const float* intr    = (const float*)d_in[2];
    const float* c2w     = (const float*)d_in[3];
    const float* ln_q_w  = (const float*)d_in[4];
    const float* ln_q_b  = (const float*)d_in[5];
    const float* Wq      = (const float*)d_in[6];
    const float* bq      = (const float*)d_in[7];
    const float* ln_k_w  = (const float*)d_in[8];
    const float* ln_k_b  = (const float*)d_in[9];
    const float* Wk      = (const float*)d_in[10];
    const float* bk      = (const float*)d_in[11];
    const float* ln_v_w  = (const float*)d_in[12];
    const float* ln_v_b  = (const float*)d_in[13];
    const float* Wv      = (const float*)d_in[14];
    const float* bv      = (const float*)d_in[15];
    const float* Wo      = (const float*)d_in[16];
    const float* bo      = (const float*)d_in[17];
    const float* ln_pre_w  = (const float*)d_in[18];
    const float* ln_pre_b  = (const float*)d_in[19];
    const float* W1      = (const float*)d_in[20];
    const float* b1      = (const float*)d_in[21];
    const float* W2      = (const float*)d_in[22];
    const float* b2      = (const float*)d_in[23];
    const float* ln_post_w = (const float*)d_in[24];
    const float* ln_post_b = (const float*)d_in[25];

    char* ws = (char*)d_ws;
    size_t cursor = 0;
    auto alloc = [&](size_t bytes) { size_t r = cursor; cursor += (bytes + 255) & ~(size_t)255; return r; };

    size_t of1 = alloc(4ull * 1024 * 1024 * 4);
    size_t of2 = alloc(4ull * 1024 * 1024 * 4);
    size_t opi = alloc(2ull * 4 * 1024 * 4 * 4);
    size_t ooi = alloc(2ull * 4 * 4 * 4);
    size_t owq = alloc(320ull * 320 * 2);
    size_t owk = alloc(320ull * 320 * 2);
    size_t owv = alloc(320ull * 320 * 2);
    size_t owo = alloc(320ull * 320 * 2);
    size_t ow1 = alloc(320ull * 640 * 2);
    size_t ow2 = alloc(640ull * 320 * 2);
    size_t oXq = alloc(4096ull * 320 * 2);
    size_t oXk = alloc(4096ull * 320 * 2);
    size_t oXv = alloc(4096ull * 320 * 2);
    size_t oqf = alloc(4096ull * 320 * 4);
    size_t okf = alloc(4096ull * 320 * 4);
    size_t ovf = alloc(4096ull * 320 * 4);
    size_t oqp = alloc(4ull * 8 * 1024 * 64 * 2);
    size_t okT = alloc(4ull * 8 * 64 * 1024 * 2);
    size_t ovp = alloc(4ull * 8 * 1024 * 48 * 2);
    // reuse dead regions
    size_t oabf = oqf;
    size_t oz0  = okf;
    size_t oz1f = ovf;
    size_t oz1b = oXq;
    size_t ofb  = oqp;
    size_t oyf  = oXk;

    epi_prep_kernel<<<8, 256, 0, stream>>>(intr, c2w, (float*)(ws + opi), (float*)(ws + ooi));
    epi_rows_kernel<<<8192, 256, 0, stream>>>((const float*)(ws + opi), (const float*)(ws + ooi),
                                              (float*)(ws + of1), (float*)(ws + of2));
    cvt_bf16_kernel<<<400, 256, 0, stream>>>(Wq, (unsigned short*)(ws + owq), 320 * 320);
    cvt_bf16_kernel<<<400, 256, 0, stream>>>(Wk, (unsigned short*)(ws + owk), 320 * 320);
    cvt_bf16_kernel<<<400, 256, 0, stream>>>(Wv, (unsigned short*)(ws + owv), 320 * 320);
    cvt_bf16_kernel<<<400, 256, 0, stream>>>(Wo, (unsigned short*)(ws + owo), 320 * 320);
    cvt_bf16_kernel<<<800, 256, 0, stream>>>(W1, (unsigned short*)(ws + ow1), 320 * 640);
    cvt_bf16_kernel<<<800, 256, 0, stream>>>(W2, (unsigned short*)(ws + ow2), 640 * 320);
    ln_kernel<<<4096, 256, 0, stream>>>(x,    ln_q_w, ln_q_b, (unsigned short*)(ws + oXq), nullptr, nullptr, 1);
    ln_kernel<<<4096, 256, 0, stream>>>(srcE, ln_k_w, ln_k_b, (unsigned short*)(ws + oXk), nullptr, nullptr, 1);
    ln_kernel<<<4096, 256, 0, stream>>>(srcE, ln_v_w, ln_v_b, (unsigned short*)(ws + oXv), nullptr, nullptr, 1);
    gemm_bf16_kernel<<<320, 128, 0, stream>>>((const unsigned int*)(ws + oXq), (const unsigned int*)(ws + owq),
                                              bq, nullptr, (float*)(ws + oqf), nullptr, 4096, 320, 320, 0);
    gemm_bf16_kernel<<<320, 128, 0, stream>>>((const unsigned int*)(ws + oXk), (const unsigned int*)(ws + owk),
                                              bk, nullptr, (float*)(ws + okf), nullptr, 4096, 320, 320, 0);
    gemm_bf16_kernel<<<320, 128, 0, stream>>>((const unsigned int*)(ws + oXv), (const unsigned int*)(ws + owv),
                                              bv, nullptr, (float*)(ws + ovf), nullptr, 4096, 320, 320, 0);
    pack_q_kernel<<<8192, 256, 0, stream>>>((const float*)(ws + oqf), (unsigned short*)(ws + oqp));
    pack_k_kernel<<<8192, 256, 0, stream>>>((const float*)(ws + okf), (unsigned short*)(ws + okT));
    pack_v_kernel<<<6144, 256, 0, stream>>>((const float*)(ws + ovf), (unsigned short*)(ws + ovp));
    attn_kernel<<<512, 128, 0, stream>>>((const unsigned int*)(ws + oqp), (const unsigned short*)(ws + okT),
                                         (const unsigned short*)(ws + ovp), (const float*)(ws + of1),
                                         (const float*)(ws + of2), (unsigned short*)(ws + oabf));
    gemm_bf16_kernel<<<320, 128, 0, stream>>>((const unsigned int*)(ws + oabf), (const unsigned int*)(ws + owo),
                                              bo, nullptr, (float*)(ws + oz0), nullptr, 4096, 320, 320, 0);
    ln_kernel<<<4096, 256, 0, stream>>>((const float*)(ws + oz0), ln_pre_w, ln_pre_b,
                                        (unsigned short*)(ws + oz1b), (float*)(ws + oz1f), nullptr, 0);
    gemm_bf16_kernel<<<640, 128, 0, stream>>>((const unsigned int*)(ws + oz1b), (const unsigned int*)(ws + ow1),
                                              b1, nullptr, nullptr, (unsigned short*)(ws + ofb), 4096, 640, 320, 1);
    gemm_bf16_kernel<<<320, 128, 0, stream>>>((const unsigned int*)(ws + ofb), (const unsigned int*)(ws + ow2),
                                              b2, (const float*)(ws + oz1f), (float*)(ws + oyf), nullptr, 4096, 320, 640, 0);
    ln_kernel<<<4096, 256, 0, stream>>>((const float*)(ws + oyf), ln_post_w, ln_post_b,
                                        nullptr, nullptr, (float*)d_out, 0);
}
